// MasifSiteProNet_42614665511297
// MI455X (gfx1250) — matrix operand order repaired and checked by reading, not compiled, not run on hardware
//
#include <hip/hip_runtime.h>

typedef __attribute__((ext_vector_type(2)))  float    v2f;
typedef __attribute__((ext_vector_type(4)))  float    v4f;
typedef __attribute__((ext_vector_type(8)))  float    v8f;
typedef __attribute__((ext_vector_type(8)))  _Float16 v8h;
typedef __attribute__((ext_vector_type(16))) _Float16 v16h;

#define B_ 8
#define V_ 8192
#define N_ 2048
#define C_ 128

// ---------------------------------------------------------------------------
// Kernel 0: one-time prep — convert W1 [k][n] (f32) to transposed f16
// W1T [n][k] so MLP B-fragments become two contiguous b128 loads per lane.
// 128x128 elements, 1 block of 256 threads, 64 elements each.
// ---------------------------------------------------------------------------
__global__ __launch_bounds__(256) void masif_w1_prep_kernel(
    const float* __restrict__ W1, _Float16* __restrict__ W1T)
{
    const int tid = threadIdx.x;
#pragma unroll
    for (int i = 0; i < 64; ++i) {
        const int linear = tid * 64 + i;        // 0 .. 16383
        const int n = linear >> 7;              // output column
        const int k = linear & 127;             // input row
        W1T[(size_t)n * C_ + k] = (_Float16)W1[(size_t)k * C_ + n];
    }
}

// ---------------------------------------------------------------------------
// Kernel 1: nearest-neighbor argmin via V_WMMA_F32_16X16X4_F32.
// Each wave32 owns 16 vertices. A = verts tile (16x4, K={x,y,z,0}); per
// 16-position chunk B = graph_pos chunk (4x16). The argmin key is
// |p|^2 - 2*dot (|v|^2 is row-constant, exact fp32 like the reference).
// Per-lane running min+index over 128 chunks, then shfl_xor min-reduction
// across each 16-lane half.
// ---------------------------------------------------------------------------
__global__ __launch_bounds__(256) void masif_argmin_kernel(
    const float* __restrict__ verts,   // [B,V,3]
    const float* __restrict__ gpos,    // [B,N,3]
    int* __restrict__ idx_out)         // [B,V]
{
    const int lane = threadIdx.x & 31;
    const int wave = threadIdx.x >> 5;
    const int wgid = blockIdx.x * 8 + wave;           // 4096 waves total
    const int b    = wgid / (V_ / 16);
    const int v0   = (wgid % (V_ / 16)) * 16;
    const int m    = lane & 15;
    const bool hi  = lane >= 16;

    const float* vr = verts + ((size_t)b * V_ + (v0 + m)) * 3;
    const float ax = vr[0], ay = vr[1], az = vr[2];
    v2f a;
    a.x = hi ? az   : ax;   // VGPR0: K0 (lo half) / K2 (hi half)
    a.y = hi ? 0.0f : ay;   // VGPR1: K1 (lo half) / K3 pad (hi half)

    float best[8];
    int   bidx[8];
#pragma unroll
    for (int r = 0; r < 8; ++r) { best[r] = __builtin_inff(); bidx[r] = 0; }

    const float* gb = gpos + (size_t)b * N_ * 3;
    for (int n0 = 0; n0 < N_; n0 += 16) {
        const float* pr = gb + (size_t)(n0 + m) * 3;
        const float px = pr[0], py = pr[1], pz = pr[2];
        const float n2 = px * px + py * py + pz * pz;   // |p|^2 for column n
        v2f bf;
        bf.x = hi ? pz   : px;
        bf.y = hi ? 0.0f : py;

        v8f acc = {};
        acc = __builtin_amdgcn_wmma_f32_16x16x4_f32(
            false, a, false, bf, (short)0, acc, false, false);

        const int n = n0 + m;
#pragma unroll
        for (int r = 0; r < 8; ++r) {
            const float score = fmaf(-2.0f, acc[r], n2);
            const bool lt = score < best[r];          // strict < keeps first hit
            best[r] = lt ? score : best[r];
            bidx[r] = lt ? n : bidx[r];
        }
    }

#pragma unroll
    for (int s = 1; s < 16; s <<= 1) {                // stays within 16-lane half
#pragma unroll
        for (int r = 0; r < 8; ++r) {
            const float ob = __shfl_xor(best[r], s, 32);
            const int   oi = __shfl_xor(bidx[r], s, 32);
            const bool take = (ob < best[r]) || (ob == best[r] && oi < bidx[r]);
            best[r] = take ? ob : best[r];
            bidx[r] = take ? oi : bidx[r];
        }
    }

    if (lane == 0 || lane == 16) {
        const int rowoff = hi ? 8 : 0;
#pragma unroll
        for (int r = 0; r < 8; ++r)
            idx_out[(size_t)b * V_ + v0 + rowoff + r] = bidx[r];
    }
}

// ---------------------------------------------------------------------------
// Kernel 2: gather + MLP via V_WMMA_F32_16X16X32_F16 (f16 in, fp32 acc).
// Each wave owns 32 vertices (two 16-row A tiles) so every W1T B-fragment
// (two b128 loads, pre-converted f16, L2-resident) feeds two WMMAs.
// Per N-tile: 4 K-chunks x 2 tiles = 8 WMMAs; relu, +b1, *W2 fused
// in-register; lane-sum reduction + b2 gives the scalar per vertex.
//
// f16 A-fragment (16x32): element e<8  -> K = kc + kbA + e,
//                         element e>=8 -> K = kc + 16 + kbA + (e-8).
// f16 B-fragment (32x16): N = lane%16, element e<8  -> K = kc + kbB + e,
//                         element e>=8 -> K = kc + 16 + kbB + (e-8).
// ---------------------------------------------------------------------------
__global__ __launch_bounds__(256) void masif_mlp_kernel(
    const float* __restrict__ processed,   // [B,N,C]
    const _Float16* __restrict__ W1T,      // [C_out][C_in] f16
    const float* __restrict__ b1,          // [C]
    const float* __restrict__ W2,          // [C,1]
    const float* __restrict__ b2,          // [1]
    const int* __restrict__ idx,           // [B,V]
    float* __restrict__ out)               // [B,V,1]
{
    const int lane = threadIdx.x & 31;
    const int wave = threadIdx.x >> 5;
    const int wgid = blockIdx.x * 8 + wave;          // 2048 waves total
    const int b    = wgid / (V_ / 32);
    const int v0   = (wgid % (V_ / 32)) * 32;
    const int m    = lane & 15;
    const bool hi  = lane >= 16;
    const int kbA  = hi ? 8 : 0;     // A K-base for this lane half
    const int kbB  = hi ? 8 : 0;     // B K-base for this lane half

    // Two gathered rows per lane (tile 0: rows v0..v0+15, tile 1: +16)
    const int idx0 = idx[(size_t)b * V_ + v0 + m];
    const int idx1 = idx[(size_t)b * V_ + v0 + 16 + m];
    const float* row0 = processed + ((size_t)b * N_ + idx0) * C_;
    const float* row1 = processed + ((size_t)b * N_ + idx1) * C_;

    // Preload A fragments for both tiles, all 4 K-chunks (vector loads + cvt)
    v16h afrag0[4], afrag1[4];
#pragma unroll
    for (int kc = 0; kc < 4; ++kc) {
        const int o0 = kc * 32 + kbA;
        const v4f a0 = *(const v4f*)(row0 + o0);
        const v4f a1 = *(const v4f*)(row0 + o0 + 4);
        const v4f a2 = *(const v4f*)(row0 + o0 + 16);
        const v4f a3 = *(const v4f*)(row0 + o0 + 20);
        const v4f c0 = *(const v4f*)(row1 + o0);
        const v4f c1 = *(const v4f*)(row1 + o0 + 4);
        const v4f c2 = *(const v4f*)(row1 + o0 + 16);
        const v4f c3 = *(const v4f*)(row1 + o0 + 20);
#pragma unroll
        for (int e = 0; e < 4; ++e) {
            afrag0[kc][e]      = (_Float16)a0[e];
            afrag0[kc][4 + e]  = (_Float16)a1[e];
            afrag0[kc][8 + e]  = (_Float16)a2[e];
            afrag0[kc][12 + e] = (_Float16)a3[e];
            afrag1[kc][e]      = (_Float16)c0[e];
            afrag1[kc][4 + e]  = (_Float16)c1[e];
            afrag1[kc][8 + e]  = (_Float16)c2[e];
            afrag1[kc][12 + e] = (_Float16)c3[e];
        }
    }

    float part0[8], part1[8];
#pragma unroll
    for (int r = 0; r < 8; ++r) { part0[r] = 0.0f; part1[r] = 0.0f; }

    for (int nt = 0; nt < 8; ++nt) {
        const int n = nt * 16 + m;          // output-feature column (this lane)
        const _Float16* bp = W1T + (size_t)n * C_ + kbB;
        v8f acc0 = {}, acc1 = {};
#pragma unroll
        for (int kc = 0; kc < 4; ++kc) {
            const v8h blo = *(const v8h*)(bp + kc * 32);       // K = kc+kbB+0..7
            const v8h bhi = *(const v8h*)(bp + kc * 32 + 16);  // K = kc+kbB+16..23
            const v16h bfrag = __builtin_shufflevector(
                blo, bhi, 0, 1, 2, 3, 4, 5, 6, 7, 8, 9, 10, 11, 12, 13, 14, 15);
            acc0 = __builtin_amdgcn_wmma_f32_16x16x32_f16(
                false, afrag0[kc], false, bfrag, (short)0, acc0, false, false);
            acc1 = __builtin_amdgcn_wmma_f32_16x16x32_f16(
                false, afrag1[kc], false, bfrag, (short)0, acc1, false, false);
        }
        const float bias = b1[n];
        const float w2n  = W2[n];
#pragma unroll
        for (int r = 0; r < 8; ++r) {
            float h0 = acc0[r] + bias;
            float h1 = acc1[r] + bias;
            h0 = h0 > 0.0f ? h0 : 0.0f;
            h1 = h1 > 0.0f ? h1 : 0.0f;
            part0[r] = fmaf(h0, w2n, part0[r]);
            part1[r] = fmaf(h1, w2n, part1[r]);
        }
    }

    // Sum-reduce across the 16 lanes of each half
#pragma unroll
    for (int s = 1; s < 16; s <<= 1) {
#pragma unroll
        for (int r = 0; r < 8; ++r) {
            part0[r] += __shfl_xor(part0[r], s, 32);
            part1[r] += __shfl_xor(part1[r], s, 32);
        }
    }

    if (lane == 0 || lane == 16) {
        const float bb = b2[0];
        const int rowoff = hi ? 8 : 0;
#pragma unroll
        for (int r = 0; r < 8; ++r) {
            out[(size_t)b * V_ + v0 + rowoff + r]      = part0[r] + bb;
            out[(size_t)b * V_ + v0 + 16 + rowoff + r] = part1[r] + bb;
        }
    }
}

extern "C" void kernel_launch(void* const* d_in, const int* in_sizes, int n_in,
                              void* d_out, int out_size, void* d_ws, size_t ws_size,
                              hipStream_t stream) {
    const float* verts     = (const float*)d_in[0];  // [B,V,3]
    const float* gpos      = (const float*)d_in[1];  // [B,N,3]
    const float* processed = (const float*)d_in[2];  // [B,N,C]
    const float* W1        = (const float*)d_in[3];  // [C,C]
    const float* b1        = (const float*)d_in[4];  // [C]
    const float* W2        = (const float*)d_in[5];  // [C,1]
    const float* b2        = (const float*)d_in[6];  // [1]
    float* out = (float*)d_out;                      // [B,V,1]

    // Workspace layout: [0, 256KB) idx (B*V ints); [256KB, +32KB) W1T f16.
    int* idx_ws = (int*)d_ws;
    _Float16* w1t = (_Float16*)((char*)d_ws + (size_t)B_ * V_ * sizeof(int));

    masif_w1_prep_kernel<<<dim3(1), dim3(256), 0, stream>>>(W1, w1t);
    masif_argmin_kernel<<<dim3(512), dim3(256), 0, stream>>>(verts, gpos, idx_ws);
    masif_mlp_kernel<<<dim3(256), dim3(256), 0, stream>>>(processed, w1t, b1, W2,
                                                          b2, idx_ws, out);
}
